// ODConv2d_26637387170431
// MI455X (gfx1250) — matrix operand order repaired and checked by reading, not compiled, not run on hardware
//
#include <hip/hip_runtime.h>
#include <hip/hip_bf16.h>
#include <math.h>

// Problem constants
#define NB   16
#define CIN  128
#define COUT 128
#define HH   64
#define WW   64
#define ATT  16

typedef __attribute__((ext_vector_type(16))) __bf16       v16bf;
typedef __attribute__((ext_vector_type(8)))  float        v8f;
typedef __attribute__((ext_vector_type(4)))  unsigned int v4u;

union Frag { v16bf bf; v4u u[2]; };

// ---------------- workspace layout (bytes) ----------------
// ch   : [NB][CIN]  f32            @ 0        (8192)
// fl   : [NB][COUT] f32            @ 8192     (8192)
// coef : [NB][9*4*9] f32           @ 16384    (20736)
// agg  : [NB][9][COUT][CIN] bf16   @ 40960    (4718592)
// xpad : [NB][66][66][CIN]  bf16   @ 4759552  (17842176)   total ~22.6 MB
#define WS_CH   0
#define WS_FL   8192
#define WS_COEF 16384
#define WS_AGG  40960
#define WS_XPAD 4759552

__device__ __forceinline__ float sigmf(float z) { return 1.f / (1.f + __expf(-z)); }

// ---------------------------------------------------------------------------
// Kernel 1: attention trunk + heads.  One block per batch sample.
// ---------------------------------------------------------------------------
__global__ void __launch_bounds__(256) attn_kernel(
    const float* __restrict__ x,    const float* __restrict__ fc_w,
    const float* __restrict__ gam,  const float* __restrict__ bet,
    const float* __restrict__ mu,   const float* __restrict__ var,
    const float* __restrict__ ch_w, const float* __restrict__ ch_b,
    const float* __restrict__ fl_w, const float* __restrict__ fl_b,
    const float* __restrict__ sp_w, const float* __restrict__ sp_b,
    const float* __restrict__ fd_w, const float* __restrict__ fd_b,
    const float* __restrict__ kr_w, const float* __restrict__ kr_b,
    float* __restrict__ chv, float* __restrict__ flv, float* __restrict__ coef) {
  const int b    = blockIdx.x;
  const int tid  = threadIdx.x;
  const int wave = tid >> 5, lane = tid & 31;

  __shared__ float spool[CIN];
  __shared__ float sh[ATT];
  __shared__ float sspa[9], sfld[9], sker[4];

  // global average pool: one wave per channel (wave32 shuffle reduce)
  for (int c = wave; c < CIN; c += 8) {
    const float* px = x + ((size_t)b * CIN + c) * (HH * WW);
    float s = 0.f;
    for (int j = lane; j < HH * WW; j += 32) s += px[j];
    #pragma unroll
    for (int off = 16; off > 0; off >>= 1) s += __shfl_xor(s, off, 32);
    if (lane == 0) spool[c] = s * (1.f / (HH * WW));
  }
  __syncthreads();

  // 1x1 conv (linear) + BN(eval) + ReLU
  if (tid < ATT) {
    float d = 0.f;
    const float* fw = fc_w + tid * CIN;
    for (int i = 0; i < CIN; ++i) d += spool[i] * fw[i];
    float sc = gam[tid] * rsqrtf(var[tid] + 1e-5f);
    float hv = (d - mu[tid]) * sc + bet[tid];
    sh[tid] = fmaxf(hv, 0.f);
  }
  __syncthreads();

  // channel / filter attention
  if (tid < CIN) {
    float d = ch_b[tid];
    const float* wv = ch_w + tid * ATT;
    #pragma unroll
    for (int j = 0; j < ATT; ++j) d += sh[j] * wv[j];
    chv[b * CIN + tid] = sigmf(d);
  } else {
    int o = tid - CIN;
    float d = fl_b[o];
    const float* wv = fl_w + o * ATT;
    #pragma unroll
    for (int j = 0; j < ATT; ++j) d += sh[j] * wv[j];
    flv[b * COUT + o] = sigmf(d);
  }

  // small heads: spatial(9), field(9), kernel softmax(4)
  if (tid < 9) {
    float d = sp_b[tid];
    const float* wv = sp_w + tid * ATT;
    #pragma unroll
    for (int j = 0; j < ATT; ++j) d += sh[j] * wv[j];
    sspa[tid] = sigmf(d);
  } else if (tid >= 16 && tid < 25) {
    int u = tid - 16;
    float d = fd_b[u];
    const float* wv = fd_w + u * ATT;
    #pragma unroll
    for (int j = 0; j < ATT; ++j) d += sh[j] * wv[j];
    sfld[u] = sigmf(d);
  } else if (tid == 32) {
    float l[4], m = -1e30f;
    #pragma unroll
    for (int n = 0; n < 4; ++n) {
      float d = kr_b[n];
      const float* wv = kr_w + n * ATT;
      #pragma unroll
      for (int j = 0; j < ATT; ++j) d += sh[j] * wv[j];
      l[n] = d; m = fmaxf(m, d);
    }
    float se = 0.f;
    #pragma unroll
    for (int n = 0; n < 4; ++n) { l[n] = __expf(l[n] - m); se += l[n]; }
    #pragma unroll
    for (int n = 0; n < 4; ++n) sker[n] = l[n] / se;
  }
  __syncthreads();

  // coef[b][pq][n][uv] = spa[pq]*ker[n]*fld[uv]   (324 values, > blockDim -> loop)
  for (int t = tid; t < 324; t += 256) {
    int pq = t / 36, r = t - pq * 36, n = r / 9, uv = r - n * 9;
    coef[b * 324 + t] = sspa[pq] * sker[n] * sfld[uv];
  }
}

// ---------------------------------------------------------------------------
// Kernel 2: padded HWC bf16 activation:  xpad[b][hp][wp][i] = x*ch  (0 border)
// ---------------------------------------------------------------------------
__global__ void __launch_bounds__(256) xpad_kernel(
    const float* __restrict__ x, const float* __restrict__ chv,
    __bf16* __restrict__ xpad) {
  int idx = blockIdx.x * 256 + threadIdx.x;           // NB*66*66*CIN = 8921088
  int i = idx & (CIN - 1);
  int t = idx >> 7;
  int wp = t % 66; t /= 66;
  int hp = t % 66;
  int b  = t / 66;
  float v = 0.f;
  if (hp >= 1 && hp <= HH && wp >= 1 && wp <= WW) {
    v = x[(((size_t)b * CIN + i) * HH + (hp - 1)) * WW + (wp - 1)] * chv[b * CIN + i];
  }
  xpad[idx] = (__bf16)v;
}

// ---------------------------------------------------------------------------
// Kernel 3: dynamic weight aggregation -> bf16 agg[b][pq][o][i]
// ---------------------------------------------------------------------------
__global__ void __launch_bounds__(256) agg_kernel(
    const float* __restrict__ weight, const float* __restrict__ coef,
    __bf16* __restrict__ agg) {
  const int b = blockIdx.z, pq = blockIdx.y;
  const int oi = blockIdx.x * 256 + threadIdx.x;
  const int o = oi >> 7, i = oi & (CIN - 1);
  __shared__ float sc[36];
  if (threadIdx.x < 36) sc[threadIdx.x] = coef[b * 324 + pq * 36 + threadIdx.x];
  __syncthreads();
  float acc = 0.f;
  #pragma unroll
  for (int n = 0; n < 4; ++n) {
    const float* wp = weight + ((((size_t)n * COUT + o) * CIN + i) * 9 + pq) * 9;
    #pragma unroll
    for (int uv = 0; uv < 9; ++uv) acc = fmaf(wp[uv], sc[n * 9 + uv], acc);
  }
  agg[(((size_t)b * 9 + pq) * COUT + o) * CIN + i] = (__bf16)acc;
}

// ---------------------------------------------------------------------------
// Kernel 4: per-sample 3x3 conv as 9 shifted bf16 WMMA GEMMs.
// Block = (b, image row h): 8 waves x (M=16 o-tile, N=64 pixels).
// 3 padded input rows (50.7 KB) staged via ASYNC global->LDS (no VGPR
// round-trip, tracked by ASYNCcnt), K-loop = 9 pq x 4 k-steps of
// v_wmma_f32_16x16x32_bf16, next A-panel prefetched during compute.
// ---------------------------------------------------------------------------
__global__ void __launch_bounds__(256) conv_wmma_kernel(
    const __bf16* __restrict__ agg, const __bf16* __restrict__ xpad,
    const float* __restrict__ flv, float* __restrict__ out) {
  const int h = blockIdx.x, b = blockIdx.y;
  const int tid = threadIdx.x;
  const int wave = tid >> 5, lane = tid & 31;
  const int hi = lane >> 4, ln = lane & 15;

  // stage padded rows h..h+2 (contiguous in xpad) into LDS via async DMA
  __shared__ __attribute__((aligned(16))) __bf16 lbs[3 * 66 * CIN];
  {
    const char* src = (const char*)(xpad + ((size_t)(b * 66 + h)) * 66 * CIN);
    // generic->LDS: low 32 bits of a generic LDS pointer are the LDS byte offset
    unsigned lbase = (unsigned)(uintptr_t)(&lbs[0]);
    const int nchunk = 3 * 66 * CIN * 2 / 16;        // 3168 x 16B
    for (int i = tid; i < nchunk; i += 256) {
      unsigned laddr = lbase + (unsigned)i * 16u;
      unsigned long long gaddr = (unsigned long long)(uintptr_t)(src + (size_t)i * 16);
      asm volatile("global_load_async_to_lds_b128 %0, %1, off"
                   :: "v"(laddr), "v"(gaddr) : "memory");
    }
    asm volatile("s_wait_asynccnt 0" ::: "memory");
  }
  __syncthreads();

  v8f acc[4] = {};                                    // 16x64 fp32 accumulators
  const __bf16* aggb = agg + (size_t)b * 9 * COUT * CIN;

  for (int pq = 0; pq < 9; ++pq) {
    const int p = pq / 3, q = pq - p * 3;
    // A row for this lane: agg[b][pq][o = wave*16 + ln][*]
    const __bf16* arow = aggb + (size_t)pq * COUT * CIN + (wave * 16 + ln) * CIN;
    // pull next pq's A panel toward the caches while this panel computes
    if (pq < 8) __builtin_prefetch((const void*)(arow + COUT * CIN), 0, 3);
    // B base: lbs[(p*66 + (w + q)) * CIN]
    const __bf16* brow = lbs + (p * 66 + q) * CIN;
    #pragma unroll
    for (int ks = 0; ks < 4; ++ks) {
      const int k0 = ks * 32;
      Frag a;                                         // 16-bit A layout: K 0-7/16-23 vs 8-15/24-31
      a.u[0] = *(const v4u*)(arow + k0 + hi * 8);
      a.u[1] = *(const v4u*)(arow + k0 + 16 + hi * 8);
      #pragma unroll
      for (int nt = 0; nt < 4; ++nt) {
        const __bf16* bp = brow + (nt * 16 + ln) * CIN + k0 + hi * 8;
        Frag bb;                                      // 16-bit B layout: K 0-7/16-23 vs 8-15/24-31
        bb.u[0] = *(const v4u*)(bp);
        bb.u[1] = *(const v4u*)(bp + 16);
        acc[nt] = __builtin_amdgcn_wmma_f32_16x16x32_bf16(
            false, a.bf, false, bb.bf, (short)0, acc[nt], false, false);
      }
    }
  }

  // scale by filter attention and store (C/D layout: lane half -> M +8, VGPR r -> M row)
  float fls[8];
  const float* fb = flv + b * COUT + wave * 16 + hi * 8;
  #pragma unroll
  for (int r = 0; r < 8; ++r) fls[r] = fb[r];
  float* ob = out + (((size_t)b * COUT + wave * 16 + hi * 8) * HH + h) * WW;
  #pragma unroll
  for (int nt = 0; nt < 4; ++nt) {
    #pragma unroll
    for (int r = 0; r < 8; ++r) {
      ob[(size_t)r * HH * WW + nt * 16 + ln] = acc[nt][r] * fls[r];
    }
  }
}

// ---------------------------------------------------------------------------
extern "C" void kernel_launch(void* const* d_in, const int* in_sizes, int n_in,
                              void* d_out, int out_size, void* d_ws, size_t ws_size,
                              hipStream_t stream) {
  (void)in_sizes; (void)n_in; (void)out_size; (void)ws_size;
  const float* x      = (const float*)d_in[0];
  const float* weight = (const float*)d_in[1];
  const float* fc_w   = (const float*)d_in[2];
  const float* bn_g   = (const float*)d_in[3];
  const float* bn_b   = (const float*)d_in[4];
  const float* bn_m   = (const float*)d_in[5];
  const float* bn_v   = (const float*)d_in[6];
  const float* ch_w   = (const float*)d_in[7];
  const float* ch_b   = (const float*)d_in[8];
  const float* fl_w   = (const float*)d_in[9];
  const float* fl_b   = (const float*)d_in[10];
  const float* sp_w   = (const float*)d_in[11];
  const float* sp_b   = (const float*)d_in[12];
  const float* fd_w   = (const float*)d_in[13];
  const float* fd_b   = (const float*)d_in[14];
  const float* kr_w   = (const float*)d_in[15];
  const float* kr_b   = (const float*)d_in[16];

  char* ws = (char*)d_ws;
  float*  chv  = (float*)(ws + WS_CH);
  float*  flv  = (float*)(ws + WS_FL);
  float*  coef = (float*)(ws + WS_COEF);
  __bf16* aggb = (__bf16*)(ws + WS_AGG);
  __bf16* xpad = (__bf16*)(ws + WS_XPAD);

  attn_kernel<<<NB, 256, 0, stream>>>(x, fc_w, bn_g, bn_b, bn_m, bn_v,
                                      ch_w, ch_b, fl_w, fl_b, sp_w, sp_b,
                                      fd_w, fd_b, kr_w, kr_b, chv, flv, coef);

  // NB*66*66*CIN / 256 = 34848 blocks
  xpad_kernel<<<34848, 256, 0, stream>>>(x, chv, xpad);

  agg_kernel<<<dim3(COUT * CIN / 256, 9, NB), 256, 0, stream>>>(weight, coef, aggb);

  conv_wmma_kernel<<<dim3(HH, NB), 256, 0, stream>>>(aggb, xpad, flv, (float*)d_out);
}
